// GCN_MLP_48636209660258
// MI455X (gfx1250) — matrix operand order repaired and checked by reading, compiled, not run on hardware
//
#include <hip/hip_runtime.h>
#include <hip/hip_bf16.h>
#include <stdint.h>

typedef __attribute__((ext_vector_type(16))) __bf16 v16bf;
typedef __attribute__((ext_vector_type(8)))  float  v8f;
typedef __attribute__((ext_vector_type(4)))  int    i32x4;
typedef __attribute__((ext_vector_type(8)))  int    i32x8;

#define SELU_SCALE 1.0507009873554805f
#define SELU_ALPHA 1.6732632423543772f

__device__ __forceinline__ float selu_f(float x) {
  return SELU_SCALE * (x > 0.f ? x : SELU_ALPHA * (__expf(x) - 1.f));
}
__device__ __forceinline__ float hswish_f(float x) {
  float t = fminf(fmaxf(x + 3.f, 0.f), 6.f);
  return x * t * (1.f / 6.f);
}

// =======================================================================
// Tensor Data Mover: async copy of a 64x64 bf16 tile (row-major, row
// length = rowlen elements) from global into LDS.  D# per ISA 08:
//   group0: count=1 | lds_addr | global_addr(57b) | type=2
//   group1: data_size=1(2B), tensor_dim0=rowlen, tensor_dim1=nrows,
//           tile_dim0=64, tile_dim1=64, dim0_stride=rowlen
// Issued by one wave; completion tracked with TENSORcnt.
// =======================================================================
__device__ __forceinline__ void tdm_load_tile64(uint32_t lds_off, const __bf16* gptr,
                                                uint32_t rowlen, uint32_t nrows) {
  uint64_t ga = (uint64_t)(uintptr_t)gptr;
  i32x4 g0;
  g0[0] = 1;                                                  // count=1 (valid)
  g0[1] = (int)lds_off;                                       // lds_addr (bytes)
  g0[2] = (int)(uint32_t)ga;                                  // global_addr[31:0]
  g0[3] = (int)((((uint32_t)(ga >> 32)) & 0x1FFFFFFu) | (2u << 30)); // addr[56:32] | type=2
  i32x8 g1;
  g1[0] = (int)0x00010000u;                                   // wg_mask=0, data_size=1 (2B)
  g1[1] = (int)((rowlen & 0xFFFFu) << 16);                    // tensor_dim0[15:0]
  g1[2] = (int)((rowlen >> 16) | ((nrows & 0xFFFFu) << 16));  // dim0[31:16] | dim1[15:0]
  g1[3] = (int)((nrows >> 16) | (64u << 16));                 // dim1[31:16] | tile_dim0=64
  g1[4] = 64;                                                 // tile_dim1=64, tile_dim2=0
  g1[5] = (int)rowlen;                                        // tensor_dim0_stride[31:0]
  g1[6] = 0;                                                  // stride[47:32] | dim1_stride lo
  g1[7] = 0;
  asm volatile("tensor_load_to_lds %0, %1" :: "s"(g0), "s"(g1) : "memory");
}

// ---------------- WMMA bf16 fragments from LDS (16x16x32) ----------------
// A (16x32): lane m = lane&15, grp = lane>>4; VGPR v -> K = v*2 + grp*16
__device__ __forceinline__ v16bf lds_A_frag(const __bf16* s, int row0, int kloc, int lane) {
  const __bf16* p = s + (row0 + (lane & 15)) * 64 + kloc + (lane >> 4) * 16;
  v16bf a;
#pragma unroll
  for (int v = 0; v < 8; ++v) {
    const int kb = v << 1;
    a[2 * v]     = p[kb];
    a[2 * v + 1] = p[kb + 1];
  }
  return a;
}
// B (32x16): lane n = lane&15; lanes 0-15 hold K=0..15 (2/VGPR), lanes 16-31 K=16..31.
// B stored transposed in LDS as [n][k] so pairs are contiguous.
__device__ __forceinline__ v16bf lds_B_frag(const __bf16* s, int n0, int kloc, int lane) {
  const __bf16* p = s + (n0 + (lane & 15)) * 64 + kloc + (lane >> 4) * 16;
  v16bf b;
#pragma unroll
  for (int v = 0; v < 8; ++v) {
    b[2 * v]     = p[2 * v];
    b[2 * v + 1] = p[2 * v + 1];
  }
  return b;
}

// =======================================================================
// WMMA GEMM: C[M,N] = A[M,K] x Bt[N,K]^T, A/Bt bf16 in global.
// 256 threads = 8 waves; block tile 64x64; wave tile 16x32 (A-frag reuse).
// TDM double-buffered LDS staging; all dims multiples of 64.
// ACT: 0=none, 1=bias+selu.  WB16: write bf16 to Cb instead of f32 to C.
// =======================================================================
template <int ACT, int WB16>
__global__ void __launch_bounds__(256)
gemm_tdm_wmma(const __bf16* __restrict__ A, const __bf16* __restrict__ Bt,
              const float* __restrict__ bias, float* __restrict__ C,
              __bf16* __restrict__ Cb, int M, int N, int K) {
  __shared__ __bf16 sA[2][64 * 64];
  __shared__ __bf16 sB[2][64 * 64];
  const int lane = threadIdx.x & 31;
  const int wave = threadIdx.x >> 5;
  const int wm = wave & 3;        // 0..3 -> M subtile
  const int wn = wave >> 2;       // 0..1 -> N subtile pair
  const int rowBase = blockIdx.y * 64;
  const int colBase = blockIdx.x * 64;
  const bool issuer = (wave == 0);

  const uint32_t offA0 = (uint32_t)(uintptr_t)&sA[0][0];
  const uint32_t offA1 = (uint32_t)(uintptr_t)&sA[1][0];
  const uint32_t offB0 = (uint32_t)(uintptr_t)&sB[0][0];
  const uint32_t offB1 = (uint32_t)(uintptr_t)&sB[1][0];

  const int nk = K >> 6;
  if (issuer) {
    tdm_load_tile64(offA0, A + (size_t)rowBase * K, (uint32_t)K, (uint32_t)M);
    tdm_load_tile64(offB0, Bt + (size_t)colBase * K, (uint32_t)K, (uint32_t)N);
  }
  v8f acc0 = {}, acc1 = {};
  for (int kt = 0; kt < nk; ++kt) {
    if (issuer) __builtin_amdgcn_s_wait_tensorcnt(0);
    __syncthreads();                       // buf[kt&1] ready for all waves
    const int buf = kt & 1;
    if (issuer && (kt + 1) < nk) {         // overlap next copy with compute
      const int k0n = (kt + 1) << 6;
      tdm_load_tile64(buf ? offA0 : offA1, A + (size_t)rowBase * K + k0n, (uint32_t)K, (uint32_t)M);
      tdm_load_tile64(buf ? offB0 : offB1, Bt + (size_t)colBase * K + k0n, (uint32_t)K, (uint32_t)N);
    }
#pragma unroll
    for (int ks = 0; ks < 2; ++ks) {
      v16bf af  = lds_A_frag(&sA[buf][0], wm * 16, ks * 32, lane);
      v16bf bf0 = lds_B_frag(&sB[buf][0], wn * 32, ks * 32, lane);
      v16bf bf1 = lds_B_frag(&sB[buf][0], wn * 32 + 16, ks * 32, lane);
      acc0 = __builtin_amdgcn_wmma_f32_16x16x32_bf16(false, af, false, bf0, (short)0, acc0, false, false);
      acc1 = __builtin_amdgcn_wmma_f32_16x16x32_bf16(false, af, false, bf1, (short)0, acc1, false, false);
    }
    __syncthreads();                       // done reading buf before it is refilled
  }

  // C/D layout: VGPR r -> m = base + r + (lane>=16 ? 8:0), n = tile + lane&15
  const int mb = rowBase + wm * 16 + (lane >> 4) * 8;
  const int n0 = colBase + wn * 32;
#pragma unroll
  for (int t = 0; t < 2; ++t) {
    const int n = n0 + t * 16 + (lane & 15);
    const float bv = (ACT == 1) ? bias[n] : 0.f;
    const v8f& a = t ? acc1 : acc0;
#pragma unroll
    for (int r = 0; r < 8; ++r) {
      float v = a[r] + bv;
      if (ACT == 1) v = selu_f(v);
      if (WB16) Cb[(size_t)(mb + r) * N + n] = (__bf16)v;
      else      C[(size_t)(mb + r) * N + n] = v;
    }
  }
}

// ---------------- dtype conversion producers ----------------
__global__ void convert_bf16_kernel(const float* __restrict__ in, __bf16* __restrict__ out, int n) {
  int i = blockIdx.x * blockDim.x + threadIdx.x;
  if (i < n) out[i] = (__bf16)in[i];
}
// w: [K][N] f32 -> wt: [N][K] bf16
__global__ void transpose_bf16_kernel(const float* __restrict__ w, __bf16* __restrict__ wt,
                                      int K, int N) {
  int idx = blockIdx.x * blockDim.x + threadIdx.x;
  if (idx >= K * N) return;
  int k = idx / N, n = idx - k * N;
  wt[(size_t)n * K + k] = (__bf16)w[idx];
}

// ---------------- CNN feature path: one workgroup per node ----------------
struct CnnParams {
  const float* w[8];
  const float* b[8];
};

__global__ void __launch_bounds__(256)
cnn_feature_kernel(const float* __restrict__ sig,
                   const float* __restrict__ w0, const float* __restrict__ b0,
                   const float* __restrict__ w1, const float* __restrict__ b1,
                   const float* __restrict__ w2, const float* __restrict__ b2,
                   CnnParams cp, float* __restrict__ hcat) {
  __shared__ float s_in[3008];
  __shared__ float s_b[4768];   // [8][596] -> [16][148] -> [32][36]
  __shared__ float s_a[4736];   // [16][296] -> [32][72]
  const int node = blockIdx.x;
  const int tid  = threadIdx.x;

  for (int i = tid; i < 3000; i += 256) s_in[i] = sig[(size_t)node * 3000 + i];
  __syncthreads();

  // fused conv0 (K=20,s=1)+selu + maxpool(k=4,s=5) -> [8][596]
  for (int idx = tid; idx < 8 * 596; idx += 256) {
    const int c = idx / 596, l = idx % 596;
    float mx = -INFINITY;
#pragma unroll
    for (int j = 0; j < 4; ++j) {
      float acc = b0[c];
      const int base = l * 5 + j;
#pragma unroll
      for (int k = 0; k < 20; ++k) acc += s_in[base + k] * w0[c * 20 + k];
      mx = fmaxf(mx, selu_f(acc));
    }
    s_b[c * 596 + l] = mx;
  }
  __syncthreads();

  // conv1: [8][596] -> [16][296], K=5, s=2, selu
  for (int idx = tid; idx < 16 * 296; idx += 256) {
    const int c = idx / 296, l = idx % 296;
    float acc = b1[c];
    for (int ci = 0; ci < 8; ++ci)
#pragma unroll
      for (int k = 0; k < 5; ++k)
        acc += s_b[ci * 596 + l * 2 + k] * w1[(c * 8 + ci) * 5 + k];
    s_a[c * 296 + l] = selu_f(acc);
  }
  __syncthreads();

  for (int idx = tid; idx < 16 * 148; idx += 256) {
    const int c = idx / 148, l = idx % 148;
    s_b[c * 148 + l] = fmaxf(s_a[c * 296 + 2 * l], s_a[c * 296 + 2 * l + 1]);
  }
  __syncthreads();

  // conv2: [16][148] -> [32][72], K=5, s=2, hardswish
  for (int idx = tid; idx < 32 * 72; idx += 256) {
    const int c = idx / 72, l = idx % 72;
    float acc = b2[c];
    for (int ci = 0; ci < 16; ++ci)
#pragma unroll
      for (int k = 0; k < 5; ++k)
        acc += s_b[ci * 148 + l * 2 + k] * w2[(c * 16 + ci) * 5 + k];
    s_a[c * 72 + l] = hswish_f(acc);
  }
  __syncthreads();

  for (int idx = tid; idx < 32 * 36; idx += 256) {
    const int c = idx / 36, l = idx % 36;
    s_b[c * 36 + l] = fmaxf(s_a[c * 72 + 2 * l], s_a[c * 72 + 2 * l + 1]);
  }
  __syncthreads();

  // 8 parallel convs + global max + tanh -> hcat[node, 512 + idx]
  const int KS[8] = {24, 20, 16, 10, 8, 6, 4, 2};
  const int ST[8] = {12, 10, 8, 5, 4, 3, 2, 1};
  for (int idx = tid; idx < 512; idx += 256) {
    const int i = idx >> 6, c = idx & 63;
    const int K = KS[i], S = ST[i];
    const int Lo = (36 - K) / S + 1;
    const float* w = cp.w[i] + (size_t)c * 32 * K;
    const float bias = cp.b[i][c];
    float mx = -INFINITY;
    for (int l = 0; l < Lo; ++l) {
      float acc = bias;
      for (int ci = 0; ci < 32; ++ci) {
        const float* wr = w + ci * K;
        const float* hr = s_b + ci * 36 + l * S;
        for (int k = 0; k < K; ++k) acc += hr[k] * wr[k];
      }
      mx = fmaxf(mx, acc);
    }
    hcat[(size_t)node * 1024 + 512 + idx] = tanhf(mx);
  }
}

// ---------------- GCN aggregation + misc elementwise ----------------
__global__ void fill_kernel(float* __restrict__ p, float v, int n) {
  int i = blockIdx.x * blockDim.x + threadIdx.x;
  if (i < n) p[i] = v;
}
__global__ void deg_edges_kernel(const int* __restrict__ dst, float* __restrict__ deg, int E) {
  int i = blockIdx.x * blockDim.x + threadIdx.x;
  if (i < E) atomicAdd(&deg[dst[i]], 1.0f);
}
__global__ void rsqrt_kernel(float* __restrict__ p, int n) {
  int i = blockIdx.x * blockDim.x + threadIdx.x;
  if (i < n) p[i] = rsqrtf(p[i]);
}
// out[n,f] = XW[n,f]*dinv[n]^2 + bias[f]
__global__ void aggr_init_kernel(const float* __restrict__ XW, const float* __restrict__ bias,
                                 const float* __restrict__ dinv, float* __restrict__ out,
                                 int F, int ldo, int total) {
  int idx = blockIdx.x * blockDim.x + threadIdx.x;
  if (idx >= total) return;
  int n = idx / F, f = idx - n * F;
  float di = dinv[n];
  out[(size_t)n * ldo + f] = XW[(size_t)idx] * di * di + bias[f];
}
__global__ void __launch_bounds__(256)
aggr_edges_kernel(const float* __restrict__ XW, const int* __restrict__ src,
                  const int* __restrict__ dst, const float* __restrict__ dinv,
                  float* __restrict__ out, int F, int ldo) {
  const int e = blockIdx.x;
  const int s = src[e], d = dst[e];
  const float nrm = dinv[s] * dinv[d];
  for (int f = threadIdx.x; f < F; f += blockDim.x)
    atomicAdd(&out[(size_t)d * ldo + f], XW[(size_t)s * F + f] * nrm);
}
// relu + bf16 copy for gemm2's A operand
__global__ void relu_bf16_kernel(const float* __restrict__ in, __bf16* __restrict__ outb, int n) {
  int i = blockIdx.x * blockDim.x + threadIdx.x;
  if (i < n) outb[i] = (__bf16)fmaxf(in[i], 0.f);
}
// cols 0..511: selu(tanh(v)); cols 512..1023: selu(v); write bf16 for fc1
__global__ void concat_act_bf16_kernel(const float* __restrict__ hcat,
                                       __bf16* __restrict__ hcatb, int total) {
  int idx = blockIdx.x * blockDim.x + threadIdx.x;
  if (idx >= total) return;
  int col = idx & 1023;
  float v = hcat[idx];
  if (col < 512) v = tanhf(v);
  hcatb[idx] = (__bf16)selu_f(v);
}
// fc3: out[n] = dot(F2[n,:256], w) + b; one wave per row
__global__ void __launch_bounds__(256)
fc3_kernel(const float* __restrict__ X, const float* __restrict__ w,
           const float* __restrict__ b, float* __restrict__ out, int Nn, int K) {
  const int row  = blockIdx.x * 8 + (threadIdx.x >> 5);
  const int lane = threadIdx.x & 31;
  if (row >= Nn) return;
  float acc = 0.f;
  for (int k = lane; k < K; k += 32) acc += X[(size_t)row * K + k] * w[k];
#pragma unroll
  for (int off = 16; off > 0; off >>= 1) acc += __shfl_xor(acc, off, 32);
  if (lane == 0) out[row] = acc + b[0];
}

// ---------------- host-side orchestration ----------------
extern "C" void kernel_launch(void* const* d_in, const int* in_sizes, int n_in,
                              void* d_out, int out_size, void* d_ws, size_t ws_size,
                              hipStream_t stream) {
  constexpr int NN = 8192, EE = 131072, DIN = 128, HID = 1024;
  (void)in_sizes; (void)n_in; (void)out_size; (void)ws_size;

  const float* x    = (const float*)d_in[0];
  const int*   ei   = (const int*)d_in[1];
  const int*   src  = ei;
  const int*   dst  = ei + EE;
  const float* sig  = (const float*)d_in[2];
  const float* w0 = (const float*)d_in[3];  const float* b0 = (const float*)d_in[4];
  const float* w1 = (const float*)d_in[5];  const float* b1 = (const float*)d_in[6];
  const float* w2 = (const float*)d_in[7];  const float* b2 = (const float*)d_in[8];
  CnnParams cp;
  for (int i = 0; i < 8; ++i) {
    cp.w[i] = (const float*)d_in[9 + i];
    cp.b[i] = (const float*)d_in[17 + i];
  }
  const float* gcn1_w = (const float*)d_in[25]; const float* gcn1_b = (const float*)d_in[26];
  const float* gcn2_w = (const float*)d_in[27]; const float* gcn2_b = (const float*)d_in[28];
  const float* fc1_w  = (const float*)d_in[29]; const float* fc1_b  = (const float*)d_in[30];
  const float* fc2_w  = (const float*)d_in[31]; const float* fc2_b  = (const float*)d_in[32];
  const float* fc3_w  = (const float*)d_in[33]; const float* fc3_b  = (const float*)d_in[34];
  float* out = (float*)d_out;

  // ---- workspace carving ----
  char* wp = (char*)d_ws;
  auto carve = [&](size_t bytes) { char* p = wp; wp += (bytes + 255) & ~(size_t)255; return p; };
  float*  dinv  = (float*)carve((size_t)NN * 4);
  float*  XW1   = (float*)carve((size_t)NN * HID * 4);   // also XW2 reuse
  float*  H1    = (float*)carve((size_t)NN * HID * 4);
  float*  Hcat  = (float*)carve((size_t)NN * HID * 4);
  __bf16* xb    = (__bf16*)carve((size_t)NN * DIN * 2);
  __bf16* w1t   = (__bf16*)carve((size_t)DIN * HID * 2);   // [1024][128]
  __bf16* H1b   = (__bf16*)carve((size_t)NN * HID * 2);
  __bf16* w2t   = (__bf16*)carve((size_t)HID * 512 * 2);   // [512][1024]
  __bf16* Hcatb = (__bf16*)carve((size_t)NN * HID * 2);
  __bf16* fc1t  = (__bf16*)carve((size_t)HID * 512 * 2);   // [512][1024]
  __bf16* F1b   = (__bf16*)carve((size_t)NN * 512 * 2);
  __bf16* fc2t  = (__bf16*)carve((size_t)512 * 256 * 2);   // [256][512]
  float*  XW2 = XW1;   // XW1 dead after aggregation 1
  float*  F2  = H1;    // H1(f32) dead after relu->H1b

  const int TB = 256;
  auto grid1 = [&](int n) { return dim3((n + TB - 1) / TB); };

  // --- degree / normalization ---
  fill_kernel<<<grid1(NN), TB, 0, stream>>>(dinv, 1.0f, NN);
  deg_edges_kernel<<<grid1(EE), TB, 0, stream>>>(dst, dinv, EE);
  rsqrt_kernel<<<grid1(NN), TB, 0, stream>>>(dinv, NN);

  // --- operand conversions (bf16, weights transposed to [N][K]) ---
  convert_bf16_kernel<<<grid1(NN * DIN), TB, 0, stream>>>(x, xb, NN * DIN);
  transpose_bf16_kernel<<<grid1(DIN * HID), TB, 0, stream>>>(gcn1_w, w1t, DIN, HID);
  transpose_bf16_kernel<<<grid1(HID * 512), TB, 0, stream>>>(gcn2_w, w2t, HID, 512);
  transpose_bf16_kernel<<<grid1(HID * 512), TB, 0, stream>>>(fc1_w, fc1t, HID, 512);
  transpose_bf16_kernel<<<grid1(512 * 256), TB, 0, stream>>>(fc2_w, fc2t, 512, 256);

  // --- CNN feature path (writes Hcat[:, 512:1024], f32) ---
  cnn_feature_kernel<<<NN, TB, 0, stream>>>(sig, w0, b0, w1, b1, w2, b2, cp, Hcat);

  // --- GCN layer 1: XW1 = x @ gcn1_w ---
  gemm_tdm_wmma<0, 0><<<dim3(HID / 64, NN / 64), TB, 0, stream>>>(
      xb, w1t, nullptr, XW1, nullptr, NN, HID, DIN);
  aggr_init_kernel<<<grid1(NN * HID), TB, 0, stream>>>(XW1, gcn1_b, dinv, H1, HID, HID, NN * HID);
  aggr_edges_kernel<<<EE, TB, 0, stream>>>(XW1, src, dst, dinv, H1, HID, HID);
  relu_bf16_kernel<<<grid1(NN * HID), TB, 0, stream>>>(H1, H1b, NN * HID);

  // --- GCN layer 2: XW2 = H1 @ gcn2_w -> aggregate into Hcat[:, :512] ---
  gemm_tdm_wmma<0, 0><<<dim3(512 / 64, NN / 64), TB, 0, stream>>>(
      H1b, w2t, nullptr, XW2, nullptr, NN, 512, HID);
  aggr_init_kernel<<<grid1(NN * 512), TB, 0, stream>>>(XW2, gcn2_b, dinv, Hcat, 512, HID, NN * 512);
  aggr_edges_kernel<<<EE, TB, 0, stream>>>(XW2, src, dst, dinv, Hcat, 512, HID);

  // --- tanh (gcn half) + selu (whole concat), to bf16 ---
  concat_act_bf16_kernel<<<grid1(NN * HID), TB, 0, stream>>>(Hcat, Hcatb, NN * HID);

  // --- fc1 (bf16 out), fc2 (f32 out), fc3 ---
  gemm_tdm_wmma<1, 1><<<dim3(512 / 64, NN / 64), TB, 0, stream>>>(
      Hcatb, fc1t, fc1_b, nullptr, F1b, NN, 512, HID);
  gemm_tdm_wmma<1, 0><<<dim3(256 / 64, NN / 64), TB, 0, stream>>>(
      F1b, fc2t, fc2_b, F2, nullptr, NN, 256, 512);
  fc3_kernel<<<(NN + 7) / 8, TB, 0, stream>>>(F2, fc3_w, fc3_b, out, NN, 256);
}
